// Head2Operator_5531917878038
// MI455X (gfx1250) — matrix operand order repaired and checked by reading, not compiled, not run on hardware
//
#include <hip/hip_runtime.h>

// ---- problem constants -----------------------------------------------------
#define DD 256                   // D
#define POS_TOTAL (4*256*256)    // B*L*L = 262144 positions
#define P 64                     // positions per workgroup
#define THREADS 256              // 8 wave32 waves

typedef __attribute__((ext_vector_type(16))) __bf16 bf16x16;
typedef __attribute__((ext_vector_type(8)))  __bf16 bf16x8;
typedef __attribute__((ext_vector_type(8)))  float  f32x8;
typedef __attribute__((ext_vector_type(4)))  float  f32x4;   // native ext-vector
typedef __attribute__((ext_vector_type(4)))  unsigned int u32x4;

union H8 { u32x4 u; bf16x8 h; };
union HS { __bf16 h; unsigned short s; };

static __device__ __forceinline__ bf16x8 ld8h(const void* p) {
  H8 t; t.u = *(const u32x4*)p; return t.h;
}
static __device__ __forceinline__ unsigned short f2b(float x) {
  HS t; t.h = (__bf16)x; return t.s;
}
static __device__ __forceinline__ bf16x16 cat(bf16x8 lo, bf16x8 hi) {
  return __builtin_shufflevector(lo, hi, 0,1,2,3,4,5,6,7,8,9,10,11,12,13,14,15);
}
static __device__ __forceinline__ f32x8 wmma_bf16(bf16x16 a, bf16x16 b, f32x8 c) {
  // (neg_a, A, neg_b, B, c_mod, C, reuse_a, reuse_b)
  return __builtin_amdgcn_wmma_f32_16x16x32_bf16(false, a, false, b, (short)0, c,
                                                 false, false);
}
// B-fragments: row n of W (contiguous bf16) = column n of W^T; lanes 0-15 hold
// K=0..7,16..23 of the 32-K chunk, lanes 16-31 hold K=8..15,24..31.
static __device__ __forceinline__ void loadB(const unsigned short* __restrict__ W,
                                             int n0, int l16, int lh, bf16x16 B[8]) {
  #pragma unroll
  for (int c = 0; c < 8; ++c) {
    const unsigned short* bp = W + (n0 + l16) * DD + c * 32 + lh * 8;
    B[c] = cat(ld8h(bp), ld8h(bp + 16));
  }
}

// ---- prologue: fp32 weights -> bf16 in workspace (once per launch) ---------
__global__ void cvt_weights(const float* __restrict__ Wr, const float* __restrict__ Ww,
                            unsigned short* __restrict__ WrB, unsigned short* __restrict__ WwB) {
  int i = blockIdx.x * 256 + threadIdx.x;         // 0 .. 196607
  if (i < 2*DD*DD) WrB[i] = f2b(Wr[i]);           // 131072 elems
  int j = i - 2*DD*DD;
  if (j >= 0 && j < DD*DD) WwB[j] = f2b(Ww[j]);   // 65536 elems
}

// ---- fused kernel: qk = X@Wr^T ; pred = relu(q)@k per position ; out = pred@Ww^T + b
__global__ __launch_bounds__(THREADS) void head2_fused(
    const float* __restrict__ p2,
    const unsigned short* __restrict__ WrB,   // [2D][D] bf16 bits, row-major
    const unsigned short* __restrict__ WwB,   // [D][D]  bf16 bits, row-major
    const float* __restrict__ b_w,
    float* __restrict__ out)
{
  // LDS layout: X/PRED 32KB | Q 32KB | KT 32KB | 64B zero page
  __shared__ __align__(16) unsigned char smem[98368];
  unsigned char* X  = smem;             // [P][256] bf16 (reused as PRED rows)
  unsigned char* Q  = smem + 32768;     // per pos: 16x16 bf16 row-major (relu'd)
  unsigned char* KT = smem + 65536;     // per pos: 16x16 bf16, k transposed
  unsigned char* ZP = smem + 98304;     // 64 zero bytes

  const int tid  = threadIdx.x;
  const int wave = tid >> 5;
  const int lane = tid & 31;
  const int l16  = lane & 15;
  const int lh   = lane >> 4;           // half-wave select
  const long pos0 = (long)blockIdx.x * P;

  // ---- phase 0: stream X tile fp32 -> bf16 LDS (NT loads: no L2 pollution) -
  {
    const f32x4* src = (const f32x4*)(p2 + pos0 * DD);
    #pragma unroll
    for (int i = 0; i < (P * DD / 4) / THREADS; ++i) {  // 16 iters
      int v = i * THREADS + tid;
      f32x4 f = __builtin_nontemporal_load(src + v);
      ushort4 h;
      h.x = f2b(f.x); h.y = f2b(f.y); h.z = f2b(f.z); h.w = f2b(f.w);
      *(ushort4*)(X + (long)v * 8) = h;
    }
    if (tid < 16) ((unsigned int*)ZP)[tid] = 0u;
  }
  __syncthreads();

  // ---- phase 1: QK = X @ Wr^T --------------------------------------------
  // pass 0: n-tiles {wave, wave+8}     -> q half (relu, row-major into Q)
  // pass 1: n-tiles {wave+16, wave+24} -> k half (transposed into KT)
  // `pass` is grid-uniform => scalar branch, no EXEC divergence.
  #pragma unroll 1
  for (int pass = 0; pass < 2; ++pass) {
    const int ntA = wave + 16 * pass;
    const int ntB = ntA + 8;
    bf16x16 B0[8], B1[8];
    loadB(WrB, ntA * 16, l16, lh, B0);
    loadB(WrB, ntB * 16, l16, lh, B1);
    #pragma unroll
    for (int mt = 0; mt < P / 16; ++mt) {
      f32x8 a0 = {0.f,0.f,0.f,0.f,0.f,0.f,0.f,0.f};
      f32x8 a1 = {0.f,0.f,0.f,0.f,0.f,0.f,0.f,0.f};
      const unsigned char* ab = X + ((mt*16 + l16) * DD + lh * 8) * 2;
      #pragma unroll
      for (int c = 0; c < 8; ++c) {               // K = 256, chunks of 32
        bf16x16 A = cat(ld8h(ab + c*64), ld8h(ab + c*64 + 32));
        a0 = wmma_bf16(A, B0[c], a0);             // A shared by both N-tiles
        a1 = wmma_bf16(A, B1[c], a1);
      }
      // C elem (M = v+8*lh, N = l16): position pl = mt*16 + v + 8*lh
      if (pass == 0) {
        unsigned char* q0 = Q + (mt*16 + 8*lh) * 512 + ntA * 32 + l16 * 2;
        unsigned char* q1 = Q + (mt*16 + 8*lh) * 512 + ntB * 32 + l16 * 2;
        #pragma unroll
        for (int v = 0; v < 8; ++v) {
          *(unsigned short*)(q0 + v * 512) = f2b(fmaxf(a0[v], 0.0f));
          *(unsigned short*)(q1 + v * 512) = f2b(fmaxf(a1[v], 0.0f));
        }
      } else {
        unsigned char* k0 = KT + (mt*16 + 8*lh) * 512 + l16 * 32 + (ntA - 16) * 2;
        unsigned char* k1 = KT + (mt*16 + 8*lh) * 512 + l16 * 32 + (ntB - 16) * 2;
        #pragma unroll
        for (int v = 0; v < 8; ++v) {
          *(unsigned short*)(k0 + v * 512) = f2b(a0[v]);
          *(unsigned short*)(k1 + v * 512) = f2b(a1[v]);
        }
      }
    }
  }
  __syncthreads();

  // ---- phase 2: pred = relu(q) @ k, one WMMA per position (K=16 pad 32) ----
  {
    const bf16x8 zh = { (__bf16)0.f,(__bf16)0.f,(__bf16)0.f,(__bf16)0.f,
                        (__bf16)0.f,(__bf16)0.f,(__bf16)0.f,(__bf16)0.f };
    #pragma unroll
    for (int j = 0; j < P / 8; ++j) {             // 8 positions per wave
      const int p = wave * (P / 8) + j;
      // A = q (16x16): lane<16 -> row l16 K0..7 ; lane>=16 -> row l16 K8..15
      const unsigned char* ap = Q + p * 512 + l16 * 32 + lh * 16;
      bf16x16 A = cat(ld8h(ap), zh);
      // B: col l16 = KT row l16; lane<16 -> K0..7, lane>=16 -> K8..15; upper 8 zeros
      const unsigned char* bp0 = KT + p * 512 + l16 * 32 + lh * 16;
      const unsigned char* bp1 = ZP;
      bf16x16 B = cat(ld8h(bp0), ld8h(bp1));
      f32x8 acc = {0.f,0.f,0.f,0.f,0.f,0.f,0.f,0.f};
      acc = wmma_bf16(A, B, acc);
      // pred.flatten: f = M*16 + N -> PRED rows (overlap dead X region)
      unsigned char* pb = X + p * 512 + (8 * lh * 16 + l16) * 2;
      #pragma unroll
      for (int v = 0; v < 8; ++v)
        *(unsigned short*)(pb + v * 32) = f2b(acc[v]);
    }
  }
  __syncthreads();

  // ---- phase 3: OUT = PRED @ Ww^T + b_w (wave owns n-tiles {wave, wave+8}) -
  {
    const int ntA = wave, ntB = wave + 8;
    bf16x16 B0[8], B1[8];
    loadB(WwB, ntA * 16, l16, lh, B0);
    loadB(WwB, ntB * 16, l16, lh, B1);
    const float bias0 = b_w[ntA * 16 + l16];
    const float bias1 = b_w[ntB * 16 + l16];
    #pragma unroll
    for (int mt = 0; mt < P / 16; ++mt) {
      f32x8 a0 = {0.f,0.f,0.f,0.f,0.f,0.f,0.f,0.f};
      f32x8 a1 = {0.f,0.f,0.f,0.f,0.f,0.f,0.f,0.f};
      const unsigned char* ab = X + ((mt*16 + l16) * DD + lh * 8) * 2;
      #pragma unroll
      for (int c = 0; c < 8; ++c) {
        bf16x16 A = cat(ld8h(ab + c*64), ld8h(ab + c*64 + 32));
        a0 = wmma_bf16(A, B0[c], a0);
        a1 = wmma_bf16(A, B1[c], a1);
      }
      float* o0 = out + (pos0 + mt*16 + 8*lh) * DD + ntA * 16 + l16;
      float* o1 = out + (pos0 + mt*16 + 8*lh) * DD + ntB * 16 + l16;
      #pragma unroll
      for (int v = 0; v < 8; ++v) {               // 64B contiguous per half-wave
        __builtin_nontemporal_store(a0[v] + bias0, o0 + v * DD);
        __builtin_nontemporal_store(a1[v] + bias1, o1 + v * DD);
      }
    }
  }
}

// ---- launch ----------------------------------------------------------------
extern "C" void kernel_launch(void* const* d_in, const int* in_sizes, int n_in,
                              void* d_out, int out_size, void* d_ws, size_t ws_size,
                              hipStream_t stream) {
  const float* p2 = (const float*)d_in[0];   // [B,L,L,D] fp32
  const float* Wr = (const float*)d_in[1];   // [2D,D]
  const float* Ww = (const float*)d_in[2];   // [D,D]
  const float* bw = (const float*)d_in[3];   // [D]
  unsigned short* WrB = (unsigned short*)d_ws;          // 131072 bf16
  unsigned short* WwB = WrB + 2 * DD * DD;              // 65536 bf16 (384KB total ws)

  cvt_weights<<<(3 * DD * DD + 255) / 256, 256, 0, stream>>>(Wr, Ww, WrB, WwB);
  head2_fused<<<POS_TOTAL / P, THREADS, 0, stream>>>(p2, WrB, WwB, bw, (float*)d_out);
}
